// Mamba2_23656679866750
// MI455X (gfx1250) — matrix operand order repaired and checked by reading, not compiled, not run on hardware
//
#include <hip/hip_runtime.h>
#include <cstdint>

// ---------------------------------------------------------------------------
// Mamba2 forward for MI455X (gfx1250, wave32, WMMA bf16 -> f32 accumulate)
// B=2, L=2048, D_MODEL=2048, D_INNER=4096, NHEADS=64, HEADDIM=64, D_STATE=64,
// CHUNK=256, CONV_DIM=4224, D_IN_PROJ=8384
// ---------------------------------------------------------------------------

typedef __attribute__((ext_vector_type(16))) __bf16 v16bf;
typedef __attribute__((ext_vector_type(8)))  float  v8f;
typedef __attribute__((ext_vector_type(4)))  unsigned int v4u;
typedef __attribute__((ext_vector_type(8)))  int v8i_t;
typedef __attribute__((ext_vector_type(4)))  int v4i_t;

union FragU { v16bf v; uint4 u[2]; };

// A-fragment (16x32 bf16): lane m=lane&15, kh=lane>>4.
// lanes<16 hold K {0..7,16..23}; lanes>=16 hold K {8..15,24..31}.
__device__ __forceinline__ v16bf ldA(const __bf16* rowp, int kbase, int kh) {
  FragU f;
  f.u[0] = *reinterpret_cast<const uint4*>(rowp + kbase + kh * 8);
  f.u[1] = *reinterpret_cast<const uint4*>(rowp + kbase + 16 + kh * 8);
  return f.v;
}
// B-fragment (32x16 bf16): lane n=lane&15 holds column n over K
// {kh*8..kh*8+7, 16+kh*8..16+kh*8+7}. rowp points at a row of B^T (i.e. K-contiguous).
__device__ __forceinline__ v16bf ldB(const __bf16* rowp, int kbase, int kh) {
  FragU f;
  f.u[0] = *reinterpret_cast<const uint4*>(rowp + kbase + kh * 8);
  f.u[1] = *reinterpret_cast<const uint4*>(rowp + kbase + 16 + kh * 8);
  return f.v;
}
__device__ __forceinline__ v8f wmma_bf16(v16bf a, v16bf b, v8f c) {
  return __builtin_amdgcn_wmma_f32_16x16x32_bf16(false, a, false, b,
                                                 (short)0, c, false, false);
}

// Async global->LDS copy of 16 bytes (per lane), tracked by ASYNCcnt.
__device__ __forceinline__ void async_copy16(const void* gptr, void* lptr) {
  unsigned loff = (unsigned)(uintptr_t)lptr;  // low 32 bits = LDS offset
  asm volatile("global_load_async_to_lds_b128 %0, %1, off" ::"v"(loff),
               "v"(gptr)
               : "memory");
}
__device__ __forceinline__ void wait_async_le(int n) {
  if (n == 0)
    asm volatile("s_wait_asynccnt 0x0" ::: "memory");
  else
    asm volatile("s_wait_asynccnt 0x3" ::: "memory");
}

// ---------------------------------------------------------------------------
// fp32 -> bf16 conversion
// ---------------------------------------------------------------------------
__global__ void k_cvt(const float* __restrict__ in, __bf16* __restrict__ out,
                      size_t n) {
  size_t i = (size_t)blockIdx.x * 256 + threadIdx.x;
  if (i < n) out[i] = (__bf16)in[i];
}

// ---------------------------------------------------------------------------
// GEMM: C[M,N] = A[M,K] (bf16, row-major) @ B[N,K]^T (bf16, row-major)
// Block tile 128x64, 8 waves (4 M x 2 N), wave tile 32x32 (2x2 WMMA tiles).
// Double-buffered LDS staging via GLOBAL_LOAD_ASYNC_TO_LDS_B128 (ASYNCcnt).
// ---------------------------------------------------------------------------
__global__ __launch_bounds__(256) void k_gemm(const __bf16* __restrict__ A,
                                              const __bf16* __restrict__ Bw,
                                              float* __restrict__ C, int M,
                                              int N, int K) {
  __shared__ __bf16 As[2][128][32];
  __shared__ __bf16 Bs[2][64][32];
  const int tid = threadIdx.x;
  const int lane = tid & 31, wv = tid >> 5;
  const int ln15 = lane & 15, kh = lane >> 4;
  const int m0 = blockIdx.y * 128, n0 = blockIdx.x * 64;
  const int wm = wv & 3, wn = wv >> 2;  // 4 x 2 wave grid
  const int arow = tid >> 1, ahalf = tid & 1;
  const int brow = tid >> 2, bq = tid & 3;
  v8f acc[2][2] = {};

  const int ngroups = K / 32;
  auto issue = [&](int i) {
    const int k0 = i * 32, bufi = i & 1;
    const __bf16* ag = &A[(size_t)(m0 + arow) * K + k0 + ahalf * 16];
    async_copy16(ag, &As[bufi][arow][ahalf * 16]);
    async_copy16(ag + 8, &As[bufi][arow][ahalf * 16 + 8]);
    const __bf16* bg = &Bw[(size_t)(n0 + brow) * K + k0 + bq * 8];
    async_copy16(bg, &Bs[bufi][brow][bq * 8]);
    if ((i + 1) * 32 < K) {  // prefetch the tile after next into L2
      __builtin_prefetch(ag + 32, 0, 1);
      __builtin_prefetch(bg + 32, 0, 1);
    }
  };
  issue(0);
  for (int i = 0; i < ngroups; ++i) {
    __syncthreads();  // buf[(i+1)&1] free for reuse by the next issue
    const int more = (i + 1 < ngroups) ? 1 : 0;
    if (more) issue(i + 1);
    wait_async_le(more ? 3 : 0);  // group i complete (in-order retirement)
    __syncthreads();              // tile i visible from all waves
    const int bufi = i & 1;
    v16bf bfr[2], afr[2];
#pragma unroll
    for (int nt = 0; nt < 2; ++nt)
      bfr[nt] = ldB(&Bs[bufi][wn * 32 + nt * 16 + ln15][0], 0, kh);
#pragma unroll
    for (int mt = 0; mt < 2; ++mt)
      afr[mt] = ldA(&As[bufi][wm * 32 + mt * 16 + ln15][0], 0, kh);
#pragma unroll
    for (int mt = 0; mt < 2; ++mt)
#pragma unroll
      for (int nt = 0; nt < 2; ++nt)
        acc[mt][nt] = wmma_bf16(afr[mt], bfr[nt], acc[mt][nt]);
  }
#pragma unroll
  for (int mt = 0; mt < 2; ++mt)
#pragma unroll
    for (int nt = 0; nt < 2; ++nt)
#pragma unroll
      for (int r = 0; r < 8; ++r) {
        int m = m0 + wm * 32 + mt * 16 + kh * 8 + r;
        int n = n0 + wn * 32 + nt * 16 + ln15;
        C[(size_t)m * N + n] = acc[mt][nt][r];
      }
}

// ---------------------------------------------------------------------------
// dt = softplus(zxbcdt[:, 8320:8384] + dt_bias)
// ---------------------------------------------------------------------------
__global__ void k_dt(const float* __restrict__ zxbcdt,
                     const float* __restrict__ dt_bias,
                     float* __restrict__ dt) {
  int idx = blockIdx.x * 256 + threadIdx.x;  // 4096*64
  if (idx >= 4096 * 64) return;
  int h = idx & 63, t = idx >> 6;
  float v = zxbcdt[(size_t)t * 8384 + 8320 + h] + dt_bias[h];
  dt[idx] = (v > 20.f) ? v : log1pf(__expf(v));
}

// ---------------------------------------------------------------------------
// Depthwise causal conv (width 4) + bias + SiLU + split into x (f32),
// x*dt (bf16), B (bf16), C (bf16).
// ---------------------------------------------------------------------------
__global__ void k_conv(const float* __restrict__ zxbcdt,
                       const float* __restrict__ convw,
                       const float* __restrict__ convb,
                       const float* __restrict__ dt, float* __restrict__ xf,
                       __bf16* __restrict__ xdt, __bf16* __restrict__ Bm,
                       __bf16* __restrict__ Cm) {
  long long idx = (long long)blockIdx.x * 256 + threadIdx.x;
  if (idx >= (long long)4096 * 4224) return;
  int ch = (int)(idx % 4224);
  int t = (int)(idx / 4224);
  int l = t & 2047;
  float acc = convb[ch];
#pragma unroll
  for (int w = 0; w < 4; ++w) {
    int ls = l - 3 + w;
    if (ls >= 0)
      acc += zxbcdt[(size_t)(t - 3 + w) * 8384 + 4096 + ch] * convw[ch * 4 + w];
  }
  float v = acc / (1.f + __expf(-acc));  // SiLU
  if (ch < 4096) {
    int h = ch >> 6;
    xf[(size_t)t * 4096 + ch] = v;
    xdt[(size_t)t * 4096 + ch] = (__bf16)(v * dt[t * 64 + h]);
  } else if (ch < 4160) {
    Bm[(size_t)t * 64 + (ch - 4096)] = (__bf16)v;
  } else {
    Cm[(size_t)t * 64 + (ch - 4160)] = (__bf16)v;
  }
}

// ---------------------------------------------------------------------------
// Per-(b,h,chunk) inclusive cumsum of dt*A along positions (wave scan).
// ---------------------------------------------------------------------------
__global__ __launch_bounds__(32) void k_cumsum(const float* __restrict__ dt,
                                               const float* __restrict__ A_log,
                                               float* __restrict__ Acum) {
  int blk = blockIdx.x;  // ((b*64+h)*8 + c)
  int c = blk & 7, h = (blk >> 3) & 63, b = blk >> 9;
  int lane = threadIdx.x;
  float Ah = -__expf(A_log[h]);
  int t0 = b * 2048 + c * 256 + lane * 8;
  float vals[8], run = 0.f;
#pragma unroll
  for (int i = 0; i < 8; ++i) {
    run += dt[(size_t)(t0 + i) * 64 + h] * Ah;
    vals[i] = run;
  }
  float incl = run;
  for (int off = 1; off < 32; off <<= 1) {
    float u = __shfl_up(incl, off, 32);
    if (lane >= off) incl += u;
  }
  float base = incl - run;
  float* out = Acum + ((size_t)(b * 64 + h)) * 2048 + c * 256 + lane * 8;
#pragma unroll
  for (int i = 0; i < 8; ++i) out[i] = base + vals[i];
}

// ---------------------------------------------------------------------------
// states[b,c,h,p,n] = sum_s B[s,n] * exp(Acum[255]-Acum[s]) * xdt[s,p]
// One block (4 waves) per (b,c,h); contraction over s=256 via WMMA.
// ---------------------------------------------------------------------------
__global__ __launch_bounds__(128) void k_states(const __bf16* __restrict__ xdt,
                                                const __bf16* __restrict__ Bm,
                                                const float* __restrict__ Acum,
                                                float* __restrict__ states) {
  const int h = blockIdx.x, c = blockIdx.y, b = blockIdx.z;
  __shared__ __bf16 Xt[64][256];  // [p][s], decayed x*dt (A operand)
  __shared__ __bf16 Bt[64][256];  // [n][s] = B^T     (B operand)
  const int tid = threadIdx.x;
  const int t0 = b * 2048 + c * 256;
  const float* Ac = Acum + ((size_t)(b * 64 + h)) * 2048 + c * 256;
  const float Alast = Ac[255];
  for (int idx = tid; idx < 64 * 256; idx += 128) {
    int f = idx & 63, s = idx >> 6;
    float d = __expf(Alast - Ac[s]);
    float xv = (float)xdt[(size_t)(t0 + s) * 4096 + h * 64 + f];
    Xt[f][s] = (__bf16)(xv * d);
    Bt[f][s] = Bm[(size_t)(t0 + s) * 64 + f];
  }
  __syncthreads();
  const int lane = tid & 31, ln15 = lane & 15, kh = lane >> 4, wv = tid >> 5;
  v8f acc[4] = {};
#pragma unroll
  for (int st = 0; st < 8; ++st) {
    v16bf bfr = ldB(&Bt[wv * 16 + ln15][0], st * 32, kh);
#pragma unroll
    for (int mt = 0; mt < 4; ++mt) {
      v16bf afr = ldA(&Xt[mt * 16 + ln15][0], st * 32, kh);
      acc[mt] = wmma_bf16(afr, bfr, acc[mt]);
    }
  }
  float* So = states + (((size_t)(b * 8 + c) * 64 + h)) * 64 * 64;
#pragma unroll
  for (int mt = 0; mt < 4; ++mt)
#pragma unroll
    for (int r = 0; r < 8; ++r) {
      int p = mt * 16 + kh * 8 + r, n = wv * 16 + ln15;
      So[p * 64 + n] = acc[mt][r];
    }
}

// ---------------------------------------------------------------------------
// Inter-chunk recurrence: prev[c] = prev[c-1]*exp(Achunk[c-1]) + states[c-1]
// ---------------------------------------------------------------------------
__global__ void k_chunkscan(const float* __restrict__ states,
                            const float* __restrict__ Acum,
                            float* __restrict__ prevs) {
  int idx = blockIdx.x * 256 + threadIdx.x;  // (b,h,p,n)
  if (idx >= 2 * 64 * 64 * 64) return;
  int n = idx & 63, p = (idx >> 6) & 63, h = (idx >> 12) & 63, b = idx >> 18;
  float prev = 0.f;
  for (int c = 0; c < 8; ++c) {
    size_t off = (((size_t)(b * 8 + c) * 64 + h) * 64 + p) * 64 + n;
    prevs[off] = prev;
    float ach = Acum[((size_t)(b * 64 + h)) * 2048 + c * 256 + 255];
    prev = prev * __expf(ach) + states[off];
  }
}

// ---------------------------------------------------------------------------
// Per-(b,c,half,h): y[l,p] = exp(Acum[l]) * (C @ prev^T)[l,p]
//                         + sum_s mask(l>=s) exp(Acum[l]-Acum[s]) (C@B^T)[l,s] xdt[s,p]
//                         + D[h]*x[l,p]
// Block = 128 threads (4 waves), each wave owns a 32-row l strip.
// Cs tile (128 rows x 128B, contiguous) staged by the Tensor Data Mover.
// ---------------------------------------------------------------------------
__global__ __launch_bounds__(128) void k_y(
    const __bf16* __restrict__ Cm, const __bf16* __restrict__ Bm,
    const __bf16* __restrict__ xdt, const float* __restrict__ prevs,
    const float* __restrict__ Acum, const float* __restrict__ xf,
    const float* __restrict__ Dv, float* __restrict__ y) {
  const int h = blockIdx.x;
  const int half = blockIdx.y & 1, c = blockIdx.y >> 1;
  const int b = blockIdx.z;
  __shared__ __bf16 Cs[128][64];   // [l][n]  (A operand, K=n contiguous)
  __shared__ __bf16 Ms[128][64];   // [l][s]  masked exp(seg)*CB (A operand)
  __shared__ __bf16 Bst[64][64];   // [s][n]  (B operand for CB^T, K=n)
  __shared__ __bf16 Xst[64][64];   // [p][s]  xdt^T (B operand, K=s)
  __shared__ __bf16 PS[64][64];    // [p][n]  prev_states (B operand, K=n)
  __shared__ float Acs[256];
  const int tid = threadIdx.x, lane = tid & 31, ln15 = lane & 15,
            kh = lane >> 4, wv = tid >> 5;
  const int t0 = b * 2048 + c * 256;
  const int lbase = half * 128;

#if defined(__gfx1250__) && __has_builtin(__builtin_amdgcn_tensor_load_to_lds)
  if (wv == 0) {
    // Tensor DMA: 128 rows x 128 bytes, row pitch 128B, global -> LDS Cs.
    uint64_t ga = (uint64_t)(uintptr_t)&Cm[(size_t)(t0 + lbase) * 64];
    unsigned la = (unsigned)(uintptr_t)&Cs[0][0];
    v4u g0;
    g0[0] = 1u;                                        // count=1
    g0[1] = la;                                        // lds_addr
    g0[2] = (unsigned)(ga & 0xFFFFFFFFu);              // global_addr lo
    g0[3] = (unsigned)((ga >> 32) & 0x01FFFFFFu) | (2u << 30);  // hi | type=2
    const unsigned td0 = 128, td1 = 128, tl0 = 128, tl1 = 128, str0 = 128;
    v8i_t g1;
    g1[0] = 0;                                   // wg_mask=0, data_size=1B
    g1[1] = (int)((td0 & 0xFFFFu) << 16);        // tensor_dim0 lo16
    g1[2] = (int)(((td0 >> 16) & 0xFFFFu) | ((td1 & 0xFFFFu) << 16));
    g1[3] = (int)(((td1 >> 16) & 0xFFFFu) | ((tl0 & 0xFFFFu) << 16));
    g1[4] = (int)(tl1 & 0xFFFFu);                // tile_dim1, tile_dim2=0
    g1[5] = (int)str0;                           // tensor_dim0_stride lo32
    g1[6] = 0;
    g1[7] = 0;
    v4i_t gz4 = {0, 0, 0, 0};
    v8i_t gz8 = {0, 0, 0, 0, 0, 0, 0, 0};
    __builtin_amdgcn_tensor_load_to_lds(g0, g1, gz4, gz4, gz8, 0);
    __builtin_amdgcn_s_wait_tensorcnt(0);
  }
#else
  for (int idx = tid; idx < 128 * 64; idx += 128) {
    int n = idx & 63, l = idx >> 6;
    Cs[l][n] = Cm[(size_t)(t0 + lbase + l) * 64 + n];
  }
#endif
  const float* ps = prevs + (((size_t)(b * 8 + c) * 64 + h)) * 64 * 64;
  for (int idx = tid; idx < 64 * 64; idx += 128)
    PS[idx >> 6][idx & 63] = (__bf16)ps[idx];
  const float* Ac = Acum + ((size_t)(b * 64 + h)) * 2048 + c * 256;
  for (int s = tid; s < 256; s += 128) Acs[s] = Ac[s];
  __syncthreads();

  v8f acc[2][4] = {};
  // --- Y_off: contract n=64 with prev_states ---
#pragma unroll
  for (int kc = 0; kc < 2; ++kc) {
    v16bf bfr[4];
#pragma unroll
    for (int nt = 0; nt < 4; ++nt)
      bfr[nt] = ldB(&PS[nt * 16 + ln15][0], kc * 32, kh);
#pragma unroll
    for (int mt = 0; mt < 2; ++mt) {
      v16bf afr = ldA(&Cs[wv * 32 + mt * 16 + ln15][0], kc * 32, kh);
#pragma unroll
      for (int nt = 0; nt < 4; ++nt)
        acc[mt][nt] = wmma_bf16(afr, bfr[nt], acc[mt][nt]);
    }
  }
  // scale Y_off rows by state_decay = exp(Acum[l])
#pragma unroll
  for (int mt = 0; mt < 2; ++mt)
#pragma unroll
    for (int r = 0; r < 8; ++r) {
      float sd = __expf(Acs[lbase + wv * 32 + mt * 16 + kh * 8 + r]);
#pragma unroll
      for (int nt = 0; nt < 4; ++nt) acc[mt][nt][r] *= sd;
    }

  // --- Y_diag: loop over s tiles of 64 ---
  for (int st = 0; st < 4; ++st) {
    __syncthreads();
    for (int idx = tid; idx < 64 * 64; idx += 128) {
      int f = idx & 63, s = idx >> 6;
      Bst[s][f] = Bm[(size_t)(t0 + st * 64 + s) * 64 + f];
      Xst[f][s] = xdt[(size_t)(t0 + st * 64 + s) * 4096 + h * 64 + f];
    }
    __syncthreads();
    // CB strip [32 l][64 s] = Cs @ Bst^T  (contract n)
    v8f cb[2][4] = {};
#pragma unroll
    for (int kc = 0; kc < 2; ++kc) {
      v16bf bfr[4];
#pragma unroll
      for (int nt = 0; nt < 4; ++nt)
        bfr[nt] = ldB(&Bst[nt * 16 + ln15][0], kc * 32, kh);
#pragma unroll
      for (int mt = 0; mt < 2; ++mt) {
        v16bf afr = ldA(&Cs[wv * 32 + mt * 16 + ln15][0], kc * 32, kh);
#pragma unroll
        for (int nt = 0; nt < 4; ++nt)
          cb[mt][nt] = wmma_bf16(afr, bfr[nt], cb[mt][nt]);
      }
    }
    // mask * exp(segment decay) -> Ms (wave-private rows)
#pragma unroll
    for (int mt = 0; mt < 2; ++mt)
#pragma unroll
      for (int r = 0; r < 8; ++r) {
        int lloc = wv * 32 + mt * 16 + kh * 8 + r;
        float al = Acs[lbase + lloc];
#pragma unroll
        for (int nt = 0; nt < 4; ++nt) {
          int spos = st * 64 + nt * 16 + ln15;
          float v = ((lbase + lloc) >= spos)
                        ? cb[mt][nt][r] * __expf(al - Acs[spos])
                        : 0.f;
          Ms[lloc][nt * 16 + ln15] = (__bf16)v;
        }
      }
    // Y += Ms @ Xst (contract s tile of 64)
#pragma unroll
    for (int kc = 0; kc < 2; ++kc) {
      v16bf bfr[4];
#pragma unroll
      for (int nt = 0; nt < 4; ++nt)
        bfr[nt] = ldB(&Xst[nt * 16 + ln15][0], kc * 32, kh);
#pragma unroll
      for (int mt = 0; mt < 2; ++mt) {
        v16bf afr = ldA(&Ms[wv * 32 + mt * 16 + ln15][0], kc * 32, kh);
#pragma unroll
        for (int nt = 0; nt < 4; ++nt)
          acc[mt][nt] = wmma_bf16(afr, bfr[nt], acc[mt][nt]);
      }
    }
  }
  // epilogue: + D[h]*x, write fp32 y
  float Dh = Dv[h];
#pragma unroll
  for (int mt = 0; mt < 2; ++mt)
#pragma unroll
    for (int r = 0; r < 8; ++r) {
      int lloc = wv * 32 + mt * 16 + kh * 8 + r;
      int token = t0 + lbase + lloc;
#pragma unroll
      for (int nt = 0; nt < 4; ++nt) {
        int p = nt * 16 + ln15;
        size_t o = (size_t)token * 4096 + h * 64 + p;
        y[o] = acc[mt][nt][r] + Dh * xf[o];
      }
    }
}

// ---------------------------------------------------------------------------
// Gate + RMSNorm: v = y*silu(z); yn = bf16(v * rsqrt(mean(v^2)+eps) * norm_w)
// One block (256 threads) per token.
// ---------------------------------------------------------------------------
__global__ __launch_bounds__(256) void k_gatenorm(const float* __restrict__ y,
                                                  const float* __restrict__ zx,
                                                  const float* __restrict__ nw,
                                                  __bf16* __restrict__ yn) {
  int t = blockIdx.x;
  __shared__ float red[8];
  float v[16], ss = 0.f;
#pragma unroll
  for (int j = 0; j < 16; ++j) {
    int i = threadIdx.x + j * 256;
    float zi = zx[(size_t)t * 8384 + i];
    float g = zi / (1.f + __expf(-zi));
    float vi = y[(size_t)t * 4096 + i] * g;
    v[j] = vi;
    ss += vi * vi;
  }
  int lane = threadIdx.x & 31, wv = threadIdx.x >> 5;
  for (int off = 16; off > 0; off >>= 1) ss += __shfl_down(ss, off, 32);
  if (lane == 0) red[wv] = ss;
  __syncthreads();
  if (threadIdx.x == 0) {
    float s = 0.f;
    for (int w2 = 0; w2 < 8; ++w2) s += red[w2];
    red[0] = s;
  }
  __syncthreads();
  float scale = rsqrtf(red[0] / 4096.f + 1e-5f);
#pragma unroll
  for (int j = 0; j < 16; ++j) {
    int i = threadIdx.x + j * 256;
    yn[(size_t)t * 4096 + i] = (__bf16)(v[j] * scale * nw[i]);
  }
}

// ---------------------------------------------------------------------------
extern "C" void kernel_launch(void* const* d_in, const int* in_sizes, int n_in,
                              void* d_out, int out_size, void* d_ws,
                              size_t ws_size, hipStream_t stream) {
  const float* u = (const float*)d_in[0];
  const float* W_in = (const float*)d_in[1];
  const float* conv_w = (const float*)d_in[2];
  const float* conv_b = (const float*)d_in[3];
  const float* dt_bias = (const float*)d_in[4];
  const float* A_log = (const float*)d_in[5];
  const float* Dv = (const float*)d_in[6];
  const float* norm_w = (const float*)d_in[7];
  const float* W_out = (const float*)d_in[8];
  float* out = (float*)d_out;

  char* ws = (char*)d_ws;
  size_t off = 0;
  auto take = [&](size_t bytes) -> void* {
    void* p = ws + off;
    off = (off + bytes + 255) & ~(size_t)255;
    return p;
  };
  float* zxbcdt = (float*)take(4096ULL * 8384 * 4);   // z | xBC_raw | dt_raw
  float* dt = (float*)take(4096ULL * 64 * 4);
  __bf16* u_bf = (__bf16*)take(4096ULL * 2048 * 2);
  __bf16* Win_bf = (__bf16*)take(8384ULL * 2048 * 2);
  __bf16* Wout_bf = (__bf16*)take(2048ULL * 4096 * 2);
  float* xf = (float*)take(4096ULL * 4096 * 4);
  __bf16* xdt = (__bf16*)take(4096ULL * 4096 * 2);
  __bf16* Bm = (__bf16*)take(4096ULL * 64 * 2);
  __bf16* Cm = (__bf16*)take(4096ULL * 64 * 2);
  float* Acum = (float*)take(2ULL * 64 * 2048 * 4);
  float* states = (float*)take(2ULL * 8 * 64 * 64 * 64 * 4);
  float* prevs = (float*)take(2ULL * 8 * 64 * 64 * 64 * 4);
  float* yb = (float*)take(4096ULL * 4096 * 4);
  __bf16* yn = (__bf16*)take(4096ULL * 4096 * 2);

  size_t nU = 4096ULL * 2048, nWin = 8384ULL * 2048, nWout = 2048ULL * 4096;
  k_cvt<<<(unsigned)((nU + 255) / 256), 256, 0, stream>>>(u, u_bf, nU);
  k_cvt<<<(unsigned)((nWin + 255) / 256), 256, 0, stream>>>(W_in, Win_bf, nWin);
  k_cvt<<<(unsigned)((nWout + 255) / 256), 256, 0, stream>>>(W_out, Wout_bf,
                                                             nWout);
  // in_proj: [4096,8384] = u_bf @ Win_bf^T
  k_gemm<<<dim3(131, 32), 256, 0, stream>>>(u_bf, Win_bf, zxbcdt, 4096, 8384,
                                            2048);
  k_dt<<<(4096 * 64) / 256, 256, 0, stream>>>(zxbcdt, dt_bias, dt);
  k_conv<<<(unsigned)((4096ULL * 4224 + 255) / 256), 256, 0, stream>>>(
      zxbcdt, conv_w, conv_b, dt, xf, xdt, Bm, Cm);
  k_cumsum<<<1024, 32, 0, stream>>>(dt, A_log, Acum);
  k_states<<<dim3(64, 8, 2), 128, 0, stream>>>(xdt, Bm, Acum, states);
  k_chunkscan<<<(2 * 64 * 64 * 64) / 256, 256, 0, stream>>>(states, Acum,
                                                            prevs);
  k_y<<<dim3(64, 16, 2), 128, 0, stream>>>(Cm, Bm, xdt, prevs, Acum, xf, Dv,
                                           yb);
  k_gatenorm<<<4096, 256, 0, stream>>>(yb, zxbcdt, norm_w, yn);
  // out_proj: [4096,2048] = yn @ Wout_bf^T
  k_gemm<<<dim3(32, 32), 256, 0, stream>>>(yn, Wout_bf, out, 4096, 2048, 4096);
}
